// Attention_22041772163328
// MI455X (gfx1250) — matrix operand order repaired and checked by reading, not compiled, not run on hardware
//
#include <hip/hip_runtime.h>

#ifndef USE_TDM
#define USE_TDM 1
#endif

// ---------------------------------------------------------------------------
// Types
// ---------------------------------------------------------------------------
typedef __attribute__((ext_vector_type(16))) __bf16 v16bf;
typedef __attribute__((ext_vector_type(8)))  float  v8f;
typedef __attribute__((ext_vector_type(4)))  unsigned int u32x4;
typedef __attribute__((ext_vector_type(8)))  unsigned int u32x8;

struct __align__(16) B16 { unsigned int x, y, z, w; };  // 16-byte POD vector

union Frag16 {
  v16bf v;
  B16   q[2];
  unsigned short u[16];
};

__device__ __forceinline__ unsigned short f2bf(float f) {
  unsigned int u = __float_as_uint(f);
  u += 0x7FFFu + ((u >> 16) & 1u);   // round to nearest even
  return (unsigned short)(u >> 16);
}

__device__ __forceinline__ v8f wmma_bf16(const Frag16& a, const Frag16& b, v8f c) {
  // D(16x16 f32) = A(16x32 bf16) x B(32x16 bf16) + C
  return __builtin_amdgcn_wmma_f32_16x16x32_bf16(
      /*neg_a=*/false, a.v, /*neg_b=*/false, b.v,
      /*c_mod=*/(short)0, c, /*reuse_a=*/false, /*reuse_b=*/false);
}

#if USE_TDM
// ---------------------------------------------------------------------------
// Tensor Data Mover: DMA a 2D bf16 tile (tw x th elements, row stride in
// elements) from global memory into LDS. Issue from ONE wave per block;
// follow with s_wait_tensorcnt + block barrier before consumers read LDS.
// D# layout per CDNA5 ISA (group0 + group1, 2D tensor,
// VADDR2/3 = NULL -> 2-operand asm form). data_size=1 (2 bytes).
// ---------------------------------------------------------------------------
__device__ __forceinline__ void tdm_load_2d_bf16(const void* g, unsigned int lds_off,
                                                 unsigned int tw, unsigned int th_,
                                                 unsigned long long stride_elems) {
  const unsigned long long ga = (unsigned long long)g;
  u32x4 g0;
  g0[0] = 1u;                                   // count=1, user descriptor
  g0[1] = lds_off;                              // lds_addr (bytes)
  g0[2] = (unsigned int)ga;                     // global_addr[31:0]
  g0[3] = (unsigned int)((ga >> 32) & 0x01FFFFFFull) | 0x80000000u;  // addr[56:32] | type=2
  u32x8 g1;
  g1[0] = 0x00010000u;                          // workgroup_mask=0, data_size=1 (2B)
  g1[1] = (tw & 0xFFFFu) << 16;                 // tensor_dim0[15:0] @bit48
  g1[2] = (tw >> 16) | ((th_ & 0xFFFFu) << 16); // tensor_dim0[31:16] | tensor_dim1[15:0]
  g1[3] = (th_ >> 16) | ((tw & 0xFFFFu) << 16); // tensor_dim1[31:16] | tile_dim0
  g1[4] = th_ & 0xFFFFu;                        // tile_dim1 (tile_dim2=0)
  g1[5] = (unsigned int)stride_elems;           // tensor_dim0_stride[31:0]
  g1[6] = (unsigned int)(stride_elems >> 32) & 0xFFFFu;  // stride[47:32] (dim1_stride=0)
  g1[7] = 0u;
  asm volatile("tensor_load_to_lds %0, %1" :: "s"(g0), "s"(g1) : "memory");
}
#endif

// ---------------------------------------------------------------------------
// Problem constants
// ---------------------------------------------------------------------------
#define BATCH   2
#define NCTX    2048
#define DIM     1024
#define HEADS   8
#define DQK     128
#define DV      192
#define QKVOUT  1344            // 8*128 + 128 + 192
#define NPAIR   256
#define DPAIR   128
#define HDV     (HEADS * DV)    // 1536
#define ROWS    (BATCH * NCTX)  // 4096
#define SCALE_Q 0.125f          // 64^-0.5
#define LN_EPS  1e-5f
#define RMS_EPS 1.1920928955078125e-07f

// ---------------------------------------------------------------------------
// fp32 -> bf16 conversion
// ---------------------------------------------------------------------------
__global__ void cvt_f32_bf16(const float* __restrict__ src,
                             unsigned short* __restrict__ dst, int n) {
  for (int i = blockIdx.x * blockDim.x + threadIdx.x; i < n;
       i += gridDim.x * blockDim.x)
    dst[i] = f2bf(src[i]);
}

// ---------------------------------------------------------------------------
// Generic bf16 GEMM: C[MxN] f32 = A[MxK] * B[KxN], row-major.
// Block tile 128x64, 8 waves (4x2), wave tile 32x32 (2x2 WMMA frags), K-step 32.
// A tile staged into LDS by the Tensor Data Mover; B tile transposed manually.
// ---------------------------------------------------------------------------
__global__ __launch_bounds__(256) void gemm_bf16(
    const unsigned short* __restrict__ A, const unsigned short* __restrict__ B,
    float* __restrict__ C, int M, int N, int K) {
  __shared__ __align__(16) unsigned short As[128 * 32];  // [row][k]
  __shared__ __align__(16) unsigned short Bt[64 * 32];   // [n][k] (transposed)

  const int tid  = threadIdx.x;
  const int wave = tid >> 5;
  const int lane = tid & 31;
  const int half = lane >> 4;
  const int l16  = lane & 15;
  const int wm   = wave >> 1;   // 0..3
  const int wn   = wave & 1;    // 0..1
  const int m0   = blockIdx.y * 128;
  const int n0   = blockIdx.x * 64;
#if USE_TDM
  const unsigned int ldsAs = (unsigned int)(unsigned long long)(size_t)As;
#endif

  v8f c[2][2];
  for (int i = 0; i < 2; ++i)
    for (int j = 0; j < 2; ++j)
      for (int e = 0; e < 8; ++e) c[i][j][e] = 0.0f;

  for (int k0 = 0; k0 < K; k0 += 32) {
    __syncthreads();
#if USE_TDM
    if (wave == 0)  // DMA A tile: 128 rows x 32 cols bf16, row stride K
      tdm_load_2d_bf16(A + (size_t)m0 * K + k0, ldsAs, 32u, 128u,
                       (unsigned long long)K);
#else
    {  // stage A: 128 rows x 32 cols; 2 threads/row, 16 bf16 (32B) each
      const int row = tid >> 1, h = tid & 1;
      const unsigned short* srcA = A + (size_t)(m0 + row) * K + k0 + h * 16;
      B16* dst = (B16*)(As + row * 32 + h * 16);
      dst[0] = ((const B16*)srcA)[0];
      dst[1] = ((const B16*)srcA)[1];
    }
#endif
    {  // stage B transposed: 32 k-rows, 8 threads/row, 8 bf16 each
      const int krow = tid >> 3, seg = tid & 7;
      const unsigned short* src = B + (size_t)(k0 + krow) * N + n0 + seg * 8;
      B16 tmp = *(const B16*)src;
      const unsigned short* t16 = (const unsigned short*)&tmp;
      for (int e = 0; e < 8; ++e) Bt[(seg * 8 + e) * 32 + krow] = t16[e];
      if (k0 + 32 < K) __builtin_prefetch(src + (size_t)32 * N, 0, 0);
    }
#if USE_TDM
    if (wave == 0) __builtin_amdgcn_s_wait_tensorcnt(0);
#endif
    __syncthreads();

    Frag16 a[2], bb[2];
    for (int i = 0; i < 2; ++i) {  // A frag: row = l16, Ks: kg..kg+7, kg+16..kg+23
      const unsigned short* ap = As + (wm * 32 + i * 16 + l16) * 32 + half * 8;
      a[i].q[0] = ((const B16*)ap)[0];
      a[i].q[1] = *(const B16*)(ap + 16);
    }
    for (int j = 0; j < 2; ++j) {  // B frag: col = l16, Ks: kg..kg+7, kg+16..kg+23
      const unsigned short* bp = Bt + (wn * 32 + j * 16 + l16) * 32 + half * 8;
      bb[j].q[0] = ((const B16*)bp)[0];
      bb[j].q[1] = *(const B16*)(bp + 16);
    }
    for (int i = 0; i < 2; ++i)
      for (int j = 0; j < 2; ++j) c[i][j] = wmma_bf16(a[i], bb[j], c[i][j]);
  }

  for (int i = 0; i < 2; ++i)
    for (int j = 0; j < 2; ++j)
      for (int v = 0; v < 8; ++v) {
        const int row = m0 + wm * 32 + i * 16 + 8 * half + v;
        const int col = n0 + wn * 32 + j * 16 + l16;
        C[(size_t)row * N + col] = c[i][j][v];
      }
}

// ---------------------------------------------------------------------------
// Per-row LayerNorm (+scale, +rotary) of qkv; writes bf16 q/k/v.
// One block per (b,n) row; 10 waves: 8 q-heads, 1 k, 1 v.
// Lane owns d = lane + 32*i  ->  rotate_half pairs (d, d+64) stay in-lane.
// ---------------------------------------------------------------------------
__global__ __launch_bounds__(320) void qkv_norm(
    const float* __restrict__ qkv, const float* __restrict__ rot,
    const float* __restrict__ qg, const float* __restrict__ kgam,
    const float* __restrict__ vgam, unsigned short* __restrict__ qb,
    unsigned short* __restrict__ kb, unsigned short* __restrict__ vb) {
  const int r = blockIdx.x;           // 0..4095
  const int b = r >> 11, n = r & 2047;
  const int wave = threadIdx.x >> 5;
  const int lane = threadIdx.x & 31;
  const float* row = qkv + (size_t)r * QKVOUT;

  if (wave < 9) {  // q heads (0..7) and k (8): length 128
    const int off = (wave < 8) ? wave * DQK : HEADS * DQK;
    const float* g = (wave < 8) ? qg : kgam;
    float t[4], s = 0.f, sq = 0.f;
    for (int i = 0; i < 4; ++i) {
      t[i] = row[off + lane + 32 * i];
      s += t[i];
      sq += t[i] * t[i];
    }
    for (int o = 1; o < 32; o <<= 1) {
      s  += __shfl_xor(s, o, 32);
      sq += __shfl_xor(sq, o, 32);
    }
    const float mean = s * (1.0f / 128.0f);
    const float var  = sq * (1.0f / 128.0f) - mean * mean;
    const float rs   = rsqrtf(var + LN_EPS);
    const float scl  = (wave < 8) ? SCALE_Q : 1.0f;
    float o4[4];
    for (int i = 0; i < 4; ++i)
      o4[i] = (t[i] - mean) * rs * g[lane + 32 * i] * scl;
    for (int i = 0; i < 4; ++i) {
      const int d = lane + 32 * i;
      const float p  = rot[(size_t)n * DQK + d];
      const float rh = (i < 2) ? -o4[i + 2] : o4[i - 2];  // rotate_half
      const float ov = o4[i] * cosf(p) + rh * sinf(p);
      if (wave < 8)
        qb[(((size_t)(b * HEADS + wave) * NCTX) + n) * DQK + d] = f2bf(ov);
      else
        kb[((size_t)b * NCTX + n) * DQK + d] = f2bf(ov);
    }
  } else {  // v: length 192, no rotary
    float t[6], s = 0.f, sq = 0.f;
    for (int i = 0; i < 6; ++i) {
      t[i] = row[HEADS * DQK + DQK + lane + 32 * i];
      s += t[i];
      sq += t[i] * t[i];
    }
    for (int o = 1; o < 32; o <<= 1) {
      s  += __shfl_xor(s, o, 32);
      sq += __shfl_xor(sq, o, 32);
    }
    const float mean = s * (1.0f / 192.0f);
    const float var  = sq * (1.0f / 192.0f) - mean * mean;
    const float rs   = rsqrtf(var + LN_EPS);
    for (int i = 0; i < 6; ++i) {
      const int d = lane + 32 * i;
      vb[((size_t)b * NCTX + n) * DV + d] =
          f2bf((t[i] - mean) * rs * vgam[d]);
    }
  }
}

// ---------------------------------------------------------------------------
// Pairwise bias: RMSNorm -> exact GELU -> Linear(128->8); one wave per row.
// Output layout bias[b][h][i][j]  (256x256 per (b,h)).
// ---------------------------------------------------------------------------
__global__ __launch_bounds__(256) void bias_kernel(
    const float* __restrict__ pw, const float* __restrict__ rmsg,
    const float* __restrict__ Wb, float* __restrict__ biasOut) {
  const int wave = threadIdx.x >> 5;
  const int lane = threadIdx.x & 31;
  const int R = blockIdx.x * 8 + wave;  // 0..131071: b*65536 + i*256 + j
  const float* pr = pw + (size_t)R * DPAIR;

  float t[4], sq = 0.f;
  for (int i = 0; i < 4; ++i) {
    t[i] = pr[lane * 4 + i];
    sq += t[i] * t[i];
  }
  for (int o = 1; o < 32; o <<= 1) sq += __shfl_xor(sq, o, 32);
  const float rs = rsqrtf(sq * (1.0f / 128.0f) + RMS_EPS);
  float g[4];
  for (int i = 0; i < 4; ++i) {
    const float x = t[i] * rs * rmsg[lane * 4 + i];
    g[i] = 0.5f * x * (1.0f + erff(x * 0.70710678118654752f));
  }
  const int b = R >> 16, ij = R & 65535;
  for (int h = 0; h < HEADS; ++h) {
    float p = 0.f;
    for (int i = 0; i < 4; ++i) p += g[i] * Wb[(lane * 4 + i) * HEADS + h];
    for (int o = 1; o < 32; o <<= 1) p += __shfl_xor(p, o, 32);
    if (lane == 0)
      biasOut[((size_t)(b * HEADS + h) << 16) + ij] = p;
  }
}

// ---------------------------------------------------------------------------
// Flash attention with WMMA. Block = 4 waves x 16 query rows = 64 rows.
// grid = (NCTX/64, HEADS, BATCH). Per 32-key step: 8 WMMAs for S, 12 for O.
// K tile staged by TDM; V tile transposed manually into LDS.
// ---------------------------------------------------------------------------
__global__ __launch_bounds__(128) void flash_attn(
    const unsigned short* __restrict__ qb, const unsigned short* __restrict__ kb,
    const unsigned short* __restrict__ vb, const float* __restrict__ bias,
    unsigned short* __restrict__ ob) {
  __shared__ __align__(16) unsigned short Ks[32 * 128];   // [key][d]
  __shared__ __align__(16) unsigned short Vt[192 * 32];   // [d][key]
  __shared__ __align__(16) unsigned short Ps[4 * 16 * 32];  // per-wave P tile

  const int tid  = threadIdx.x;
  const int wave = tid >> 5;
  const int lane = tid & 31;
  const int half = lane >> 4;
  const int l16  = lane & 15;
  const int b = blockIdx.z, h = blockIdx.y;
  const int q0 = blockIdx.x * 64 + wave * 16;
#if USE_TDM
  const unsigned int ldsKs = (unsigned int)(unsigned long long)(size_t)Ks;
#endif

  // Q fragments (16 rows x 128 d), bf16, loaded once.
  Frag16 aq[4];
  const unsigned short* qrow =
      qb + (((size_t)(b * HEADS + h) * NCTX) + q0 + l16) * DQK;
  for (int kc = 0; kc < 4; ++kc) {
    const unsigned short* qp = qrow + kc * 32 + half * 16;
    aq[kc].q[0] = ((const B16*)qp)[0];
    aq[kc].q[1] = ((const B16*)qp)[1];
  }

  float mrow[8], lrow[8];
  v8f acc[12];
  for (int v = 0; v < 8; ++v) { mrow[v] = -3.0e38f; lrow[v] = 0.f; }
  for (int f = 0; f < 12; ++f)
    for (int e = 0; e < 8; ++e) acc[f][e] = 0.f;

  const unsigned short* Kb = kb + (size_t)b * NCTX * DQK;
  const unsigned short* Vb = vb + (size_t)b * NCTX * DV;
  const float* biasbh = bias + (size_t)(b * HEADS + h) * NPAIR * NPAIR;

  for (int j0 = 0; j0 < NCTX; j0 += 32) {
    __syncthreads();
#if USE_TDM
    if (wave == 0)  // DMA K tile: 32 keys x 128 d bf16, contiguous rows
      tdm_load_2d_bf16(Kb + (size_t)j0 * DQK, ldsKs, (unsigned int)DQK, 32u,
                       (unsigned long long)DQK);
#else
    {  // stage K: 32 keys x 128 d; 4 threads/key, 32 bf16 (64B) each
      const int key = tid >> 2, seg = tid & 3;
      const B16* src = (const B16*)(Kb + (size_t)(j0 + key) * DQK + seg * 32);
      B16* dst = (B16*)(Ks + key * 128 + seg * 32);
      dst[0] = src[0]; dst[1] = src[1]; dst[2] = src[2]; dst[3] = src[3];
    }
#endif
    // stage V transposed: Vt[d][key]
    for (int idx = tid; idx < 32 * DV; idx += 128) {
      const int key = idx / DV, d = idx % DV;
      Vt[d * 32 + key] = Vb[(size_t)(j0 + key) * DV + d];
    }
#if USE_TDM
    if (wave == 0) __builtin_amdgcn_s_wait_tensorcnt(0);
#endif
    __syncthreads();

    // S = Q * K^T : two 16x16 f32 frags covering keys [j0, j0+32)
    v8f s0, s1;
    for (int e = 0; e < 8; ++e) { s0[e] = 0.f; s1[e] = 0.f; }
    for (int kc = 0; kc < 4; ++kc) {
      Frag16 bk0, bk1;
      const unsigned short* kp0 = Ks + l16 * 128 + kc * 32 + half * 16;
      const unsigned short* kp1 = Ks + (16 + l16) * 128 + kc * 32 + half * 16;
      bk0.q[0] = ((const B16*)kp0)[0]; bk0.q[1] = ((const B16*)kp0)[1];
      bk1.q[0] = ((const B16*)kp1)[0]; bk1.q[1] = ((const B16*)kp1)[1];
      s0 = wmma_bf16(aq[kc], bk0, s0);
      s1 = wmma_bf16(aq[kc], bk1, s1);
    }

    // bias + softclamp + online softmax (row = q0 + 8*half + v, col = l16)
    const int jb0 = (j0 + l16) >> 3, jb1 = (j0 + 16 + l16) >> 3;
    for (int v = 0; v < 8; ++v) {
      const int qi = q0 + 8 * half + v;
      const int ib = qi >> 3;
      const float b0 = biasbh[ib * NPAIR + jb0];
      const float b1 = biasbh[ib * NPAIR + jb1];
      const float x0 = 5.0f * tanhf((s0[v] + b0) * 0.2f);
      const float x1 = 5.0f * tanhf((s1[v] + b1) * 0.2f);
      float mx = fmaxf(x0, x1);
      for (int o = 1; o < 16; o <<= 1) mx = fmaxf(mx, __shfl_xor(mx, o, 32));
      const float mn    = fmaxf(mrow[v], mx);
      const float alpha = __expf(mrow[v] - mn);
      const float p0 = __expf(x0 - mn);
      const float p1 = __expf(x1 - mn);
      float ls = p0 + p1;
      for (int o = 1; o < 16; o <<= 1) ls += __shfl_xor(ls, o, 32);
      lrow[v] = lrow[v] * alpha + ls;
      mrow[v] = mn;
      for (int f = 0; f < 12; ++f) acc[f][v] *= alpha;
      const int prow = 8 * half + v;
      Ps[wave * 512 + prow * 32 + l16]      = f2bf(p0);
      Ps[wave * 512 + prow * 32 + 16 + l16] = f2bf(p1);
    }
    // Same-wave LDS ops are in-order (ISA 7.3): P store->load needs no barrier.

    // O += P * V : P as A-frag (16x32), V as 12 B-frags (32x16)
    Frag16 pa;
    const unsigned short* pp = Ps + wave * 512 + l16 * 32 + half * 16;
    pa.q[0] = ((const B16*)pp)[0];
    pa.q[1] = ((const B16*)pp)[1];
    for (int f = 0; f < 12; ++f) {
      Frag16 bv;
      const unsigned short* vp = Vt + (f * 16 + l16) * 32 + half * 16;
      bv.q[0] = ((const B16*)vp)[0];
      bv.q[1] = ((const B16*)vp)[1];
      acc[f] = wmma_bf16(pa, bv, acc[f]);
    }
  }

  // epilogue: normalize and write [b][n][h*192 + d] bf16
  for (int v = 0; v < 8; ++v) {
    const float inv = 1.0f / lrow[v];
    const int row = q0 + 8 * half + v;
    for (int f = 0; f < 12; ++f)
      ob[((size_t)b * NCTX + row) * HDV + h * DV + f * 16 + l16] =
          f2bf(acc[f][v] * inv);
  }
}

// ---------------------------------------------------------------------------
// Host launcher
// ---------------------------------------------------------------------------
extern "C" void kernel_launch(void* const* d_in, const int* in_sizes, int n_in,
                              void* d_out, int out_size, void* d_ws,
                              size_t ws_size, hipStream_t stream) {
  (void)in_sizes; (void)n_in; (void)out_size; (void)ws_size;
  const float* x       = (const float*)d_in[0];
  const float* pairw   = (const float*)d_in[1];
  const float* rotary  = (const float*)d_in[2];
  const float* Wqkv    = (const float*)d_in[3];
  const float* Wout    = (const float*)d_in[4];
  const float* qg      = (const float*)d_in[5];
  const float* kg      = (const float*)d_in[6];
  const float* vg      = (const float*)d_in[7];
  const float* rmsg    = (const float*)d_in[8];
  const float* Wb      = (const float*)d_in[9];
  float* out = (float*)d_out;

  char* ws = (char*)d_ws;
  size_t off = 0;
  auto alloc = [&](size_t bytes) -> void* {
    void* p = ws + off;
    off += (bytes + 255) & ~(size_t)255;
    return p;
  };
  unsigned short* xb    = (unsigned short*)alloc((size_t)ROWS * DIM * 2);
  unsigned short* wqkvb = (unsigned short*)alloc((size_t)DIM * QKVOUT * 2);
  unsigned short* woutb = (unsigned short*)alloc((size_t)HDV * DIM * 2);
  float*          qkv   = (float*)alloc((size_t)ROWS * QKVOUT * 4);
  unsigned short* qb    = (unsigned short*)alloc((size_t)BATCH * HEADS * NCTX * DQK * 2);
  unsigned short* kbuf  = (unsigned short*)alloc((size_t)BATCH * NCTX * DQK * 2);
  unsigned short* vbuf  = (unsigned short*)alloc((size_t)BATCH * NCTX * DV * 2);
  float*          bias  = (float*)alloc((size_t)BATCH * HEADS * NPAIR * NPAIR * 4);
  unsigned short* ob    = (unsigned short*)alloc((size_t)ROWS * HDV * 2);

  const int nx = ROWS * DIM, nw1 = DIM * QKVOUT, nw2 = HDV * DIM;
  cvt_f32_bf16<<<(nx + 255) / 256, 256, 0, stream>>>(x, xb, nx);
  cvt_f32_bf16<<<(nw1 + 255) / 256, 256, 0, stream>>>(Wqkv, wqkvb, nw1);
  cvt_f32_bf16<<<(nw2 + 255) / 256, 256, 0, stream>>>(Wout, woutb, nw2);

  gemm_bf16<<<dim3(QKVOUT / 64, ROWS / 128), 256, 0, stream>>>(
      xb, wqkvb, qkv, ROWS, QKVOUT, DIM);

  qkv_norm<<<ROWS, 320, 0, stream>>>(qkv, rotary, qg, kg, vg, qb, kbuf, vbuf);

  bias_kernel<<<(BATCH * NPAIR * NPAIR) / 8, 256, 0, stream>>>(
      pairw, rmsg, Wb, bias);

  flash_attn<<<dim3(NCTX / 64, HEADS, BATCH), 128, 0, stream>>>(
      qb, kbuf, vbuf, bias, ob);

  gemm_bf16<<<dim3(DIM / 64, ROWS / 128), 256, 0, stream>>>(
      ob, woutb, out, ROWS, DIM, HDV);
}
